// QuantumAttentionWithQPEAndTunneling_13709535609637
// MI455X (gfx1250) — matrix operand order repaired and checked by reading, not compiled, not run on hardware
//
#include <hip/hip_runtime.h>
#include <hip/hip_bf16.h>

// ---------------------------------------------------------------------------
// QuantumAttentionWithQPEAndTunneling for MI455X (gfx1250, wave32, WMMA)
//
// Insight: the circuit after angle-embedding is a fixed 16x16 complex unitary
// per (qkv, split) -- only 48 matrices. psi_final = U * psi0(x) where psi0 is
// a product state from 4 angles. Batched over 16384 tokens this is a
// [16384 x 32] * [32 x 32] real GEMM -> v_wmma_f32_16x16x32_f16 (K=32 exact).
// Attention scores / output GEMMs (K=192 / K=1024) also map onto 16x16x32
// f16 WMMA tiles. Irregular work (gate sim, expvals, softmax, LN) stays VALU.
// ---------------------------------------------------------------------------

typedef _Float16 f16;
typedef __attribute__((ext_vector_type(16))) _Float16 v16h;
typedef __attribute__((ext_vector_type(8)))  float    v8f;

union V16H { v16h v; _Float16 h[16]; };

#define PI_F    3.14159265358979f
#define PI_2F   1.57079632679490f
#define PI_4F   0.78539816339745f
#define PI_8F   0.39269908169872f

// ---- sizes ----
#define NB   16
#define NS   1024
#define NF   64
#define NSPL 16
#define DFEAT 192                   // 3*NF
#define NTOK (NB*NS)                // 16384
#define OSIZE (NTOK*DFEAT)          // o elements in d_out
// workspace layout (halves)
#define MT_HALVES   (48*1024)
#define QKV_HALVES  (NTOK*DFEAT)    // 3145728 halves each

// ===========================================================================
// Gate helpers: 16-complex state in registers, qubit w -> bit (3-w)
// ===========================================================================
__device__ __forceinline__ void g_rx(float re[16], float im[16], int w, float t){
  float sh, ch; __sincosf(0.5f*t, &sh, &ch);
  const int m = 1 << (3-w);
  #pragma unroll
  for (int i0=0;i0<16;i0++){
    if (i0 & m) continue;
    const int i1 = i0 | m;
    float r0=re[i0], q0=im[i0], r1=re[i1], q1=im[i1];
    re[i0] = ch*r0 + sh*q1;  im[i0] = ch*q0 - sh*r1;
    re[i1] = ch*r1 + sh*q0;  im[i1] = ch*q1 - sh*r0;
  }
}
__device__ __forceinline__ void g_ry(float re[16], float im[16], int w, float t){
  float sh, ch; __sincosf(0.5f*t, &sh, &ch);
  const int m = 1 << (3-w);
  #pragma unroll
  for (int i0=0;i0<16;i0++){
    if (i0 & m) continue;
    const int i1 = i0 | m;
    float r0=re[i0], q0=im[i0], r1=re[i1], q1=im[i1];
    re[i0] = ch*r0 - sh*r1;  im[i0] = ch*q0 - sh*q1;
    re[i1] = sh*r0 + ch*r1;  im[i1] = sh*q0 + ch*q1;
  }
}
__device__ __forceinline__ void g_rz(float re[16], float im[16], int w, float t){
  float sh, ch; __sincosf(0.5f*t, &sh, &ch);
  const int m = 1 << (3-w);
  #pragma unroll
  for (int i=0;i<16;i++){
    float r=re[i], q=im[i];
    if (i & m){ re[i] = ch*r - sh*q; im[i] = ch*q + sh*r; }
    else      { re[i] = ch*r + sh*q; im[i] = ch*q - sh*r; }
  }
}
__device__ __forceinline__ void g_crx(float re[16], float im[16], int c, int w, float t){
  float sh, ch; __sincosf(0.5f*t, &sh, &ch);
  const int mc = 1 << (3-c), m = 1 << (3-w);
  #pragma unroll
  for (int i0=0;i0<16;i0++){
    if (!(i0 & mc) || (i0 & m)) continue;
    const int i1 = i0 | m;
    float r0=re[i0], q0=im[i0], r1=re[i1], q1=im[i1];
    re[i0] = ch*r0 + sh*q1;  im[i0] = ch*q0 - sh*r1;
    re[i1] = ch*r1 + sh*q0;  im[i1] = ch*q1 - sh*r0;
  }
}
__device__ __forceinline__ void g_h(float re[16], float im[16], int w){
  const float r2 = 0.70710678118655f;
  const int m = 1 << (3-w);
  #pragma unroll
  for (int i0=0;i0<16;i0++){
    if (i0 & m) continue;
    const int i1 = i0 | m;
    float r0=re[i0], q0=im[i0], r1=re[i1], q1=im[i1];
    re[i0] = r2*(r0+r1); im[i0] = r2*(q0+q1);
    re[i1] = r2*(r0-r1); im[i1] = r2*(q0-q1);
  }
}
__device__ __forceinline__ void g_iswap(float re[16], float im[16], int a, int b){
  const int ma = 1 << (3-a), mb = 1 << (3-b);
  #pragma unroll
  for (int i=0;i<16;i++){
    if ((i & ma) || !(i & mb)) continue;
    const int j = (i ^ ma) ^ mb;
    float r0=re[i], q0=im[i], r1=re[j], q1=im[j];
    re[i] = -q1; im[i] = r1;      // new|01> = i*old|10>
    re[j] = -q0; im[j] = r0;      // new|10> = i*old|01>
  }
}
__device__ __forceinline__ void g_swap(float re[16], float im[16], int a, int b){
  const int ma = 1 << (3-a), mb = 1 << (3-b);
  #pragma unroll
  for (int i=0;i<16;i++){
    if ((i & ma) || !(i & mb)) continue;
    const int j = (i ^ ma) ^ mb;
    float r0=re[i], q0=im[i];
    re[i]=re[j]; im[i]=im[j]; re[j]=r0; im[j]=q0;
  }
}
__device__ __forceinline__ void g_cp(float re[16], float im[16], int a, int b, float phi){
  float sp, cp; __sincosf(phi, &sp, &cp);
  const int ma = 1 << (3-a), mb = 1 << (3-b);
  #pragma unroll
  for (int i=0;i<16;i++){
    if (!(i & ma) || !(i & mb)) continue;
    float r=re[i], q=im[i];
    re[i] = cp*r - sp*q; im[i] = cp*q + sp*r;
  }
}
__device__ __forceinline__ void g_toff(float re[16], float im[16]){
  // Toffoli on qubits (0,1,2): |110?> <-> |111?>  i.e. idx 12<->14, 13<->15
  #pragma unroll
  for (int q3=0;q3<2;q3++){
    const int i = 12 + q3, j = 14 + q3;
    float r=re[i], q=im[i];
    re[i]=re[j]; im[i]=im[j]; re[j]=r; im[j]=q;
  }
}

// ===========================================================================
// Kernel A: build the 48 per-(qkv,split) real-stacked 32x32 matrices, f16,
// stored row-major Barr[n][k] = Mreal[n][k] (exactly the WMMA-B gather order).
// Thread t evolves basis column e_t through all 3 layers.
// ===========================================================================
__global__ void build_unitaries(const float* __restrict__ w,
                                const float* __restrict__ ew,
                                const float* __restrict__ tw,
                                f16* __restrict__ MT){
  const int pair = blockIdx.x;       // qkv*16 + split
  const int col  = threadIdx.x;      // 0..15
  float re[16], im[16];
  #pragma unroll
  for (int i=0;i<16;i++){ re[i] = (i==col)?1.f:0.f; im[i]=0.f; }

  const float* wp = w  + pair*12;
  const float* ep = ew + pair*12;
  const float* tp = tw + pair*12;

  for (int L=0; L<3; L++){
    #pragma unroll
    for (int i=0;i<4;i++){
      const float th = wp[L*4+i];
      g_rx(re,im,i,th); g_ry(re,im,i,th); g_rz(re,im,i,th);
    }
    #pragma unroll
    for (int i=0;i<4;i++){
      const float th = ep[L*4+i];
      g_crx(re,im,i,(i+2)&3,th);
      g_crx(re,im,i,(i+3)&3,th);
    }
    g_iswap(re,im,0,1); g_iswap(re,im,1,2); g_iswap(re,im,2,3); g_iswap(re,im,3,0);
    g_toff(re,im);
    g_crx(re,im,0,1,tp[L*4+0]); g_crx(re,im,1,2,tp[L*4+1]);
    g_crx(re,im,2,3,tp[L*4+2]); g_crx(re,im,3,0,tp[L*4+3]);
    // QPE forward
    #pragma unroll
    for (int i=0;i<4;i++) g_h(re,im,i);
    g_cp(re,im,0,1, PI_2F);
    g_cp(re,im,0,2, PI_4F); g_cp(re,im,1,2, PI_2F);
    g_cp(re,im,0,3, PI_8F); g_cp(re,im,1,3, PI_4F); g_cp(re,im,2,3, PI_2F);
    g_swap(re,im,0,3); g_swap(re,im,1,2);
    g_h(re,im,0);
    g_h(re,im,1); g_cp(re,im,0,1,-PI_2F);
    g_h(re,im,2); g_cp(re,im,0,2,-PI_4F); g_cp(re,im,1,2,-PI_2F);
    g_h(re,im,3); g_cp(re,im,0,3,-PI_8F); g_cp(re,im,1,3,-PI_4F); g_cp(re,im,2,3,-PI_2F);
  }

  // Mreal = [[Ur,-Ui],[Ui,Ur]]; this thread holds U column `col`
  f16* B = MT + pair*1024;
  #pragma unroll
  for (int r=0;r<16;r++){
    B[r*32 + col]          = (f16)re[r];
    B[r*32 + col + 16]     = (f16)(-im[r]);
    B[(16+r)*32 + col]     = (f16)im[r];
    B[(16+r)*32 + col+16]  = (f16)re[r];
  }
}

// ===========================================================================
// Kernel B: per wave: 16 tokens. psi0 (product state) -> LDS -> WMMA with the
// (qkv,split) matrix -> psi_final -> expvals -> q_h / k_h / Vt (f16).
// ===========================================================================
__global__ void __launch_bounds__(256)
qkv_eval(const float* __restrict__ x, const f16* __restrict__ MT,
         f16* __restrict__ q_h, f16* __restrict__ k_h, f16* __restrict__ vt){
  __shared__ _Float16 sA[8][16][32];   // per-wave psi0 tiles (row-major)
  __shared__ float    sO[8][16][32];   // per-wave psi_final tiles

  const int wv   = threadIdx.x >> 5;
  const int lane = threadIdx.x & 31;
  const int lm   = lane & 15, hh = lane >> 4;

  const int blk   = blockIdx.x;
  const int tb    = blk & 127;       // 128 token-blocks per pair
  const int pair  = blk >> 7;        // 0..47
  const int qkv   = pair >> 4;
  const int split = pair & 15;
  const int token0 = tb*128 + wv*16;

  // --- stage 1: lanes 0..15 build psi0 for one token each ---
  if (hh == 0) {
    const int t = token0 + lm;
    const float* xp = x + (size_t)t*NF + split*4;
    float s0,c0,s1,c1,s2,c2,s3,c3;
    __sincosf(0.5f*xp[0], &s0, &c0);
    __sincosf(0.5f*xp[1], &s1, &c1);
    __sincosf(0.5f*xp[2], &s2, &c2);
    __sincosf(0.5f*xp[3], &s3, &c3);
    #pragma unroll
    for (int k=0;k<16;k++){
      const float m = ((k&8)?s0:c0)*((k&4)?s1:c1)*((k&2)?s2:c2)*((k&1)?s3:c3);
      const int p = __popc(k);                     // (-i)^p factor
      const float rr = (p&1) ? 0.f : ((p&2) ? -m : m);
      const float ii = (p&1) ? ((p&2) ? m : -m) : 0.f;
      sA[wv][lm][k]    = (_Float16)rr;
      sA[wv][lm][16+k] = (_Float16)ii;
    }
  }
  __syncthreads();

  // --- stage 2: whole wave loads A/B operands, two WMMAs (re & im tiles) ---
  V16H a;
  const uint4* arow = (const uint4*)&sA[wv][lm][0];
  ((uint4*)a.h)[0] = arow[hh];        // K = hh*8 + 0..7
  ((uint4*)a.h)[1] = arow[2 + hh];    // K = 16 + hh*8 + 0..7

  const _Float16* Bp = MT + pair*1024;
  V16H b0, b1;
  {
    const uint4* p0 = (const uint4*)(Bp + lm*32 + hh*8);
    ((uint4*)b0.h)[0] = p0[0]; ((uint4*)b0.h)[1] = p0[2];
    const uint4* p1 = (const uint4*)(Bp + (16+lm)*32 + hh*8);
    ((uint4*)b1.h)[0] = p1[0]; ((uint4*)b1.h)[1] = p1[2];
  }
  v8f acc0 = {}, acc1 = {};
  acc0 = __builtin_amdgcn_wmma_f32_16x16x32_f16(false, a.v, false, b0.v, (short)0, acc0, false, false);
  acc1 = __builtin_amdgcn_wmma_f32_16x16x32_f16(false, a.v, false, b1.v, (short)0, acc1, false, false);

  #pragma unroll
  for (int r=0;r<8;r++){
    const int tok = r + 8*hh;
    sO[wv][tok][lm]      = acc0[r];   // re
    sO[wv][tok][16+lm]   = acc1[r];   // im
  }
  __syncthreads();

  // --- stage 3: lanes 0..15 compute 12 expectation values for their token ---
  if (hh == 0) {
    const int t = token0 + lm;
    float pr[16], pq[16];
    #pragma unroll
    for (int k=0;k<16;k++){ pr[k]=sO[wv][lm][k]; pq[k]=sO[wv][lm][16+k]; }
    float outv[12];
    #pragma unroll
    for (int wq=0;wq<4;wq++){
      const int m = 1 << (3-wq);
      float z=0.f, xx=0.f, yy=0.f;
      #pragma unroll
      for (int i0=0;i0<16;i0++){
        if (i0 & m) continue;
        const int i1 = i0 | m;
        z  += pr[i0]*pr[i0]+pq[i0]*pq[i0] - pr[i1]*pr[i1]-pq[i1]*pq[i1];
        xx += pr[i0]*pr[i1] + pq[i0]*pq[i1];
        yy += pr[i0]*pq[i1] - pq[i0]*pr[i1];
      }
      outv[wq] = z; outv[4+wq] = 2.f*xx; outv[8+wq] = 2.f*yy;
    }
    const int fb = split*12;
    if (qkv == 0) {
      f16* o = q_h + (size_t)t*DFEAT + fb;
      #pragma unroll
      for (int e=0;e<12;e++) o[e] = (f16)outv[e];
    } else if (qkv == 1) {
      f16* o = k_h + (size_t)t*DFEAT + fb;
      #pragma unroll
      for (int e=0;e<12;e++) o[e] = (f16)outv[e];
    } else {
      const int b = t >> 10, s = t & 1023;          // V transposed: [b][feat][tok]
      f16* o = vt + ((size_t)b*DFEAT + fb)*NS + s;
      #pragma unroll
      for (int e=0;e<12;e++) o[(size_t)e*NS] = (f16)outv[e];
    }
  }
}

// ===========================================================================
// Kernel C: scores = Q K^T / 2. One wave -> eight 16x16 tiles, 6 WMMAs each (K=192).
// ===========================================================================
__global__ void __launch_bounds__(256)
scores_kernel(const f16* __restrict__ q_h, const f16* __restrict__ k_h,
              float* __restrict__ attn){
  const int lane = threadIdx.x & 31, wv = threadIdx.x >> 5;
  const int blk = blockIdx.x;                  // 16*64
  const int qt = blk & 63, b = blk >> 6;
  const int lm = lane & 15, hh = lane >> 4;

  // A operand (Q tile rows), loaded once, reused across 8 key tiles
  const f16* qrow = q_h + ((size_t)(b*NS + qt*16 + lm))*DFEAT;
  V16H A[6];
  #pragma unroll
  for (int kc=0;kc<6;kc++){
    ((uint4*)A[kc].h)[0] = *(const uint4*)(qrow + kc*32 + hh*8);
    ((uint4*)A[kc].h)[1] = *(const uint4*)(qrow + kc*32 + 16 + hh*8);
  }

  for (int kt8 = 0; kt8 < 8; kt8++){
    const int ktile = wv*8 + kt8;
    const f16* krow = k_h + ((size_t)(b*NS + ktile*16 + lm))*DFEAT;
    v8f c = {};
    #pragma unroll
    for (int kc=0;kc<6;kc++){
      V16H B;
      const uint4* p = (const uint4*)(krow + kc*32 + hh*8);
      ((uint4*)B.h)[0] = p[0]; ((uint4*)B.h)[1] = p[2];
      c = __builtin_amdgcn_wmma_f32_16x16x32_f16(false, A[kc].v, false, B.v, (short)0, c, false, false);
    }
    float* orow = attn + ((size_t)(b*NS + qt*16))*NS + ktile*16 + lm;
    #pragma unroll
    for (int r=0;r<8;r++)
      orow[(size_t)(r + 8*hh)*NS] = 0.5f * c[r];   // /sqrt(NQ)=2
  }
}

// ===========================================================================
// Kernel D: row softmax in-place in d_out attn region.
// ===========================================================================
__global__ void __launch_bounds__(256)
softmax_kernel(float* __restrict__ attn){
  __shared__ float red[256];
  const int row = blockIdx.x, t = threadIdx.x;
  float* p = attn + (size_t)row*NS;
  float v[4];
  float mx = -3.4e38f;
  #pragma unroll
  for (int i=0;i<4;i++){ v[i] = p[t + i*256]; mx = fmaxf(mx, v[i]); }
  red[t] = mx; __syncthreads();
  for (int s=128; s>0; s>>=1){ if (t<s) red[t] = fmaxf(red[t], red[t+s]); __syncthreads(); }
  mx = red[0]; __syncthreads();
  float sum = 0.f;
  #pragma unroll
  for (int i=0;i<4;i++){ v[i] = __expf(v[i]-mx); sum += v[i]; }
  red[t] = sum; __syncthreads();
  for (int s=128; s>0; s>>=1){ if (t<s) red[t] += red[t+s]; __syncthreads(); }
  const float inv = 1.0f / red[0];
  #pragma unroll
  for (int i=0;i<4;i++) p[t + i*256] = v[i]*inv;
}

// ===========================================================================
// Kernel E: o = attn @ V. One wave -> 16x16 output tile, 32 WMMAs (K=1024).
// A = f32 attn converted in-register to f16; B = contiguous V^T halves.
// ===========================================================================
__global__ void __launch_bounds__(256)
out_gemm(const float* __restrict__ attn, const f16* __restrict__ vt,
         float* __restrict__ o){
  const int lane = threadIdx.x & 31, wv = threadIdx.x >> 5;
  const int task = blockIdx.x*8 + wv;       // 16*64*12 tasks
  const int nt = task % 12;
  const int rem = task / 12;
  const int qt = rem & 63, b = rem >> 6;
  const int lm = lane & 15, hh = lane >> 4;

  const float* arow = attn + ((size_t)(b*NS + qt*16 + lm))*NS;
  const f16*   brow = vt   + ((size_t)(b*DFEAT + nt*16 + lm))*NS;

  v8f c = {};
  for (int kc=0;kc<32;kc++){
    // A: K pairs {hh*8+0..7, 16+hh*8+0..7} from f32 attn row
    V16H A;
    const float4* pa0 = (const float4*)(arow + kc*32 + hh*8);
    const float4* pa1 = (const float4*)(arow + kc*32 + 16 + hh*8);
    float4 u0 = pa0[0], u1 = pa0[1], u2 = pa1[0], u3 = pa1[1];
    float fa[16] = {u0.x,u0.y,u0.z,u0.w, u1.x,u1.y,u1.z,u1.w,
                    u2.x,u2.y,u2.z,u2.w, u3.x,u3.y,u3.z,u3.w};
    #pragma unroll
    for (int j=0;j<16;j++) A.h[j] = (_Float16)fa[j];
    // B: K pairs {hh*8+0..7, 16+hh*8+0..7} of V^T
    V16H B;
    const uint4* pb = (const uint4*)(brow + kc*32 + hh*8);
    ((uint4*)B.h)[0] = pb[0]; ((uint4*)B.h)[1] = pb[2];
    c = __builtin_amdgcn_wmma_f32_16x16x32_f16(false, A.v, false, B.v, (short)0, c, false, false);
  }
  float* orow = o + ((size_t)(b*NS + qt*16))*DFEAT + nt*16 + lm;
  #pragma unroll
  for (int r=0;r<8;r++)
    orow[(size_t)(r + 8*hh)*DFEAT] = c[r];
}

// ===========================================================================
// Kernel F: layernorm over 192 features, in-place in d_out o region.
// ===========================================================================
__global__ void __launch_bounds__(192)
layernorm_kernel(float* __restrict__ o, const float* __restrict__ g,
                 const float* __restrict__ bt){
  __shared__ float t1[192], t2[192];
  __shared__ float r1[64], r2[64];
  const int row = blockIdx.x, t = threadIdx.x;
  const float v = o[(size_t)row*DFEAT + t];
  t1[t] = v; t2[t] = v*v;
  __syncthreads();
  if (t < 64){ r1[t] = t1[t]+t1[t+64]+t1[t+128]; r2[t] = t2[t]+t2[t+64]+t2[t+128]; }
  __syncthreads();
  for (int s=32; s>0; s>>=1){
    if (t < s){ r1[t]+=r1[t+s]; r2[t]+=r2[t+s]; }
    __syncthreads();
  }
  const float mean = r1[0] * (1.0f/192.0f);
  const float var  = r2[0] * (1.0f/192.0f) - mean*mean;
  const float nv = (v - mean) * rsqrtf(var + 1e-5f);
  o[(size_t)row*DFEAT + t] = nv * g[t] + bt[t];
}

// ===========================================================================
extern "C" void kernel_launch(void* const* d_in, const int* in_sizes, int n_in,
                              void* d_out, int out_size, void* d_ws, size_t ws_size,
                              hipStream_t stream) {
  const float* x     = (const float*)d_in[0];
  const float* w     = (const float*)d_in[1];
  const float* ew    = (const float*)d_in[2];
  const float* tw    = (const float*)d_in[3];
  const float* gamma = (const float*)d_in[4];
  const float* beta  = (const float*)d_in[5];

  float* out   = (float*)d_out;
  float* o_buf = out;                 // [16][1024][192]
  float* attn  = out + OSIZE;         // [16][1024][1024]

  // workspace: MT (48x32x32 f16) | q_h | k_h | Vt  (~18.2 MB total)
  char* ws = (char*)d_ws;
  f16* MT  = (f16*)ws;
  f16* q_h = (f16*)(ws + MT_HALVES*sizeof(f16));
  f16* k_h = q_h + QKV_HALVES;
  f16* vt  = k_h + QKV_HALVES;

  build_unitaries<<<48, 16, 0, stream>>>(w, ew, tw, MT);
  qkv_eval<<<48*128, 256, 0, stream>>>(x, MT, q_h, k_h, vt);
  scores_kernel<<<NB*64, 256, 0, stream>>>(q_h, k_h, attn);
  softmax_kernel<<<NB*NS, 256, 0, stream>>>(attn);
  out_gemm<<<(NB*64*12)/8, 256, 0, stream>>>(attn, vt, o_buf);
  layernorm_kernel<<<NB*NS, 192, 0, stream>>>(o_buf, gamma, beta);
}
